// MultiHeadAttention_76184129896978
// MI455X (gfx1250) — compile-verified, matrix operand order checked by reading, not run on hardware
//
#include <hip/hip_runtime.h>
#include <hip/hip_bf16.h>

// ---------------------------------------------------------------------------
// MI455X (gfx1250) multi-head attention forward.
// f16 storage + f32 WMMA accumulation via v_wmma_f32_16x16x32_f16 (wave32).
// Global->LDS staging uses CDNA5 async-LDS DMA (ASYNCcnt) with double buffer.
// ---------------------------------------------------------------------------

typedef __attribute__((ext_vector_type(16))) _Float16 v16h;
typedef __attribute__((ext_vector_type(8)))  _Float16 v8h;
typedef __attribute__((ext_vector_type(8)))  float    v8f;

union HFrag { v16h v; v8h h[2]; };

// Load one 16x32 f16 A/B fragment from a row-major (ld halves) tile.
// Per ISA 7.12.2: lane L holds row (L&15); K halves are
//   [off..off+7] in VGPR0-3 and [off+16..off+23] in VGPR4-7, off=(L>=16)*8.
// B fragments use the identical layout on the N-major (transposed) operand.
__device__ __forceinline__ v16h load_frag16(const _Float16* __restrict__ base,
                                            int ld, int lane) {
  const _Float16* p = base + (lane & 15) * ld + ((lane >> 4) << 3);
  HFrag f;
  f.h[0] = *(const v8h*)(p);
  f.h[1] = *(const v8h*)(p + 16);
  return f.v;
}

__device__ __forceinline__ v8f wmma16(v16h a, v16h b, v8f c) {
  return __builtin_amdgcn_wmma_f32_16x16x32_f16(false, a, false, b,
                                                (short)0, c, false, false);
}

// CDNA5 async global->LDS copy, 16B per lane (GLOBAL_LOAD_ASYNC_TO_LDS_B128,
// tracked with ASYNCcnt). LDS destination is the low 32 address bits of the
// generic shared-memory pointer (aperture mapping, ISA 10.2).
__device__ __forceinline__ void async_copy_b128(void* lds_dst,
                                                const _Float16* gsrc) {
  unsigned int lds_off = (unsigned int)(unsigned long long)(uintptr_t)lds_dst;
  unsigned long long addr = (unsigned long long)(uintptr_t)gsrc;
  asm volatile("global_load_async_to_lds_b128 %0, %1, off"
               :: "v"(lds_off), "v"(addr)
               : "memory");
}

__device__ __forceinline__ void wait_async0() {
  asm volatile("s_wait_asynccnt 0x0" ::: "memory");
}

// Problem constants (from the reference).
constexpr int   kB     = 2;
constexpr int   kSeq   = 2048;
constexpr int   kDim   = 1024;
constexpr int   kHeads = 16;
constexpr int   kHd    = 64;
constexpr int   kM     = kB * kSeq;      // 4096 rows
constexpr float kScale = 0.125f;         // 64^-0.5

// ---------------------------------------------------------------------------
// Elementwise f32 -> f16
// ---------------------------------------------------------------------------
__global__ void k_f32_to_f16(const float* __restrict__ in,
                             _Float16* __restrict__ out, int n) {
  int i = blockIdx.x * blockDim.x + threadIdx.x;
  if (i < n) out[i] = (_Float16)in[i];
}

// Transpose + convert: in f32 [R][C] -> out f16 [C][R]  (weights -> N-major)
__global__ void k_transpose_f16(const float* __restrict__ in,
                                _Float16* __restrict__ out, int R, int C) {
  int i = blockIdx.x * blockDim.x + threadIdx.x;
  if (i < R * C) {
    int r = i / C, c = i % C;
    out[(size_t)c * R + r] = (_Float16)in[i];
  }
}

// ---------------------------------------------------------------------------
// QKV projection: C[4096,3072] = Xh[4096,1024] @ Wqkv; scatter to q/k/v
// in [B,H,N,Dh] f16 layout. Block tile 128x128, K-step 32, 8 waves (4x2).
// Async double-buffered LDS staging: DMA tile kt+1 while WMMAs chew tile kt.
// ---------------------------------------------------------------------------
__global__ __launch_bounds__(256) void k_qkv_gemm(
    const _Float16* __restrict__ A,    // [4096,1024]
    const _Float16* __restrict__ Bt,   // [3072,1024]  (Wqkv^T, N-major)
    const float* __restrict__ bias,    // [3072]
    _Float16* __restrict__ qh, _Float16* __restrict__ kh,
    _Float16* __restrict__ vh) {
  __shared__ __align__(16) _Float16 sA[2][128 * 32];
  __shared__ __align__(16) _Float16 sB[2][128 * 32];
  const int tid = threadIdx.x, lane = tid & 31, w = tid >> 5;
  const int wm = w & 3, wn = w >> 2;
  const int m0 = blockIdx.x * 128, n0 = blockIdx.y * 128;
  constexpr int KT = kDim / 32;

  auto stage = [&](int kt, int buf) {
#pragma unroll
    for (int c = tid; c < 512; c += 256) {   // 128 rows x 32 halves per tile
      const int r = c >> 2, kc = (c & 3) << 3;
      async_copy_b128(&sA[buf][r * 32 + kc],
                      &A[(size_t)(m0 + r) * kDim + kt * 32 + kc]);
      async_copy_b128(&sB[buf][r * 32 + kc],
                      &Bt[(size_t)(n0 + r) * kDim + kt * 32 + kc]);
    }
  };

  v8f acc[2][4];
  v8f z = {};
#pragma unroll
  for (int i = 0; i < 2; ++i)
#pragma unroll
    for (int j = 0; j < 4; ++j) acc[i][j] = z;

  stage(0, 0);
  wait_async0();
  __syncthreads();

  for (int kt = 0; kt < KT; ++kt) {
    const int cur = kt & 1;
    if (kt + 1 < KT) stage(kt + 1, cur ^ 1);   // overlap DMA with compute

    v16h af[2], bf[4];
#pragma unroll
    for (int i = 0; i < 2; ++i)
      af[i] = load_frag16(&sA[cur][(wm * 32 + i * 16) * 32], 32, lane);
#pragma unroll
    for (int j = 0; j < 4; ++j)
      bf[j] = load_frag16(&sB[cur][(wn * 64 + j * 16) * 32], 32, lane);
#pragma unroll
    for (int i = 0; i < 2; ++i)
#pragma unroll
      for (int j = 0; j < 4; ++j) acc[i][j] = wmma16(af[i], bf[j], acc[i][j]);

    wait_async0();
    __syncthreads();
  }

  // Epilogue: C layout = VGPR e holds row (e + 8*(lane>=16)), col (lane&15).
  const int rl = (lane >> 4) << 3;
#pragma unroll
  for (int i = 0; i < 2; ++i) {
#pragma unroll
    for (int j = 0; j < 4; ++j) {
      const int gn = n0 + wn * 64 + j * 16 + (lane & 15);
      const float bn = bias[gn];
      const int which = gn >> 10;            // 0=q 1=k 2=v
      const int cc = gn & 1023;
      const int hh = cc >> 6, dd = cc & 63;
      _Float16* dst = (which == 0) ? qh : (which == 1) ? kh : vh;
#pragma unroll
      for (int e = 0; e < 8; ++e) {
        const int gm = m0 + wm * 32 + i * 16 + rl + e;
        const int bb = gm >> 11, tt = gm & 2047;
        dst[(((size_t)(bb * kHeads + hh)) * kSeq + tt) * kHd + dd] =
            (_Float16)(acc[i][j][e] + bn);
      }
    }
  }
}

// ---------------------------------------------------------------------------
// Flash attention: one WG per (b,h, 128-query block); 8 waves x 16 rows.
// K tile staged via async-LDS DMA (rows are already the Bt layout for Q@K^T);
// V staged transposed through VGPRs (Vt rows are Bt layout for P@V).
// Online softmax in f32 registers.
// ---------------------------------------------------------------------------
__global__ __launch_bounds__(256) void k_attn(
    const _Float16* __restrict__ qh, const _Float16* __restrict__ kh,
    const _Float16* __restrict__ vh, _Float16* __restrict__ oh) {
  __shared__ __align__(16) _Float16 sK[64 * 64];        // [key][dh]
  __shared__ __align__(16) _Float16 sVt[64 * 64];       // [dh][key]
  __shared__ __align__(16) _Float16 sP[8 * 16 * 64];    // per-wave P slab

  const int tid = threadIdx.x, lane = tid & 31, w = tid >> 5;
  const int bh = blockIdx.y;                     // 0..B*H-1
  const int b = bh >> 4, h = bh & 15;
  const int q0 = blockIdx.x * 128 + w * 16;      // this wave's query rows
  const size_t bhBase = (size_t)bh * kSeq * kHd;

  // Q fragments stay resident in registers for the whole pass.
  v16h qf[2];
  {
    const _Float16* qp = qh + bhBase + (size_t)q0 * kHd;
#pragma unroll
    for (int j = 0; j < 2; ++j) qf[j] = load_frag16(qp + j * 32, kHd, lane);
  }

  v8f o[4];
  v8f z = {};
#pragma unroll
  for (int c = 0; c < 4; ++c) o[c] = z;
  float mrow[8], lrow[8];
#pragma unroll
  for (int i = 0; i < 8; ++i) { mrow[i] = -1e30f; lrow[i] = 0.f; }

  _Float16* pw = &sP[w * (16 * 64)];
  const int rl = (lane >> 4) << 3;

  for (int kt = 0; kt < kSeq / 64; ++kt) {
    __syncthreads();
    const int kb = kt * 64;
#pragma unroll
    for (int c = tid; c < 512; c += 256) {       // 64 keys x 64 dh
      const int key = c >> 3, dc = (c & 7) << 3;
      // K tile: async DMA straight into LDS.
      async_copy_b128(&sK[key * 64 + dc],
                      &kh[bhBase + (size_t)(kb + key) * kHd + dc]);
      // V tile: through VGPRs (transposed on the way into LDS).
      v8h vv = *(const v8h*)&vh[bhBase + (size_t)(kb + key) * kHd + dc];
#pragma unroll
      for (int e = 0; e < 8; ++e) sVt[(dc + e) * 64 + key] = vv[e];
      if (kt + 1 < kSeq / 64) {
        __builtin_prefetch(&kh[bhBase + (size_t)(kb + 64 + key) * kHd + dc], 0, 1);
        __builtin_prefetch(&vh[bhBase + (size_t)(kb + 64 + key) * kHd + dc], 0, 1);
      }
    }
    wait_async0();
    __syncthreads();

    // S = Q @ K^T  (16 x 64 per wave)
    v8f s[4];
#pragma unroll
    for (int c = 0; c < 4; ++c) s[c] = z;
#pragma unroll
    for (int j = 0; j < 2; ++j)
#pragma unroll
      for (int c = 0; c < 4; ++c) {
        v16h kf = load_frag16(&sK[(c * 16) * 64 + j * 32], 64, lane);
        s[c] = wmma16(qf[j], kf, s[c]);
      }

    // Online softmax. Row i lives in lanes {g*16..g*16+15} at vector idx i.
#pragma unroll
    for (int i = 0; i < 8; ++i) {
      float mx = fmaxf(fmaxf(s[0][i], s[1][i]), fmaxf(s[2][i], s[3][i]));
#pragma unroll
      for (int d = 1; d < 16; d <<= 1) mx = fmaxf(mx, __shfl_xor(mx, d, 16));
      const float mnew = fmaxf(mrow[i], mx * kScale);
      const float alpha = __expf(mrow[i] - mnew);
      float rsum = 0.f;
#pragma unroll
      for (int c = 0; c < 4; ++c) {
        float p = __expf(s[c][i] * kScale - mnew);
        s[c][i] = p;
        rsum += p;
      }
#pragma unroll
      for (int d = 1; d < 16; d <<= 1) rsum += __shfl_xor(rsum, d, 16);
      lrow[i] = lrow[i] * alpha + rsum;
      mrow[i] = mnew;
#pragma unroll
      for (int c = 0; c < 4; ++c) o[c][i] *= alpha;
    }

    // C-layout -> A-fragment layout for P via wave-private LDS slab.
#pragma unroll
    for (int i = 0; i < 8; ++i)
#pragma unroll
      for (int c = 0; c < 4; ++c)
        pw[(rl + i) * 64 + c * 16 + (lane & 15)] = (_Float16)s[c][i];

    v16h pf[2];
#pragma unroll
    for (int j = 0; j < 2; ++j) pf[j] = load_frag16(pw + j * 32, 64, lane);

    // O += P @ V
#pragma unroll
    for (int j = 0; j < 2; ++j)
#pragma unroll
      for (int c = 0; c < 4; ++c) {
        v16h vf = load_frag16(&sVt[(c * 16) * 64 + j * 32], 64, lane);
        o[c] = wmma16(pf[j], vf, o[c]);
      }
  }

  // Normalize and write O in [b, token, h*64+dh] (row-major for out-proj).
#pragma unroll
  for (int i = 0; i < 8; ++i) {
    const float inv = 1.f / lrow[i];
    const int token = q0 + rl + i;
#pragma unroll
    for (int c = 0; c < 4; ++c) {
      const int dh = c * 16 + (lane & 15);
      oh[((size_t)(b * kSeq + token)) * kDim + h * kHd + dh] =
          (_Float16)(o[c][i] * inv);
    }
  }
}

// ---------------------------------------------------------------------------
// Output projection: out[4096,1024] = Oh @ Wout + b_out (f32 result).
// Same async double-buffered pipeline as the QKV GEMM.
// ---------------------------------------------------------------------------
__global__ __launch_bounds__(256) void k_out_gemm(
    const _Float16* __restrict__ A,    // [4096,1024]
    const _Float16* __restrict__ Bt,   // [1024,1024] (Wout^T)
    const float* __restrict__ bias,    // [1024]
    float* __restrict__ C) {
  __shared__ __align__(16) _Float16 sA[2][128 * 32];
  __shared__ __align__(16) _Float16 sB[2][128 * 32];
  const int tid = threadIdx.x, lane = tid & 31, w = tid >> 5;
  const int wm = w & 3, wn = w >> 2;
  const int m0 = blockIdx.x * 128, n0 = blockIdx.y * 128;
  constexpr int KT = kDim / 32;

  auto stage = [&](int kt, int buf) {
#pragma unroll
    for (int c = tid; c < 512; c += 256) {
      const int r = c >> 2, kc = (c & 3) << 3;
      async_copy_b128(&sA[buf][r * 32 + kc],
                      &A[(size_t)(m0 + r) * kDim + kt * 32 + kc]);
      async_copy_b128(&sB[buf][r * 32 + kc],
                      &Bt[(size_t)(n0 + r) * kDim + kt * 32 + kc]);
    }
  };

  v8f acc[2][4];
  v8f z = {};
#pragma unroll
  for (int i = 0; i < 2; ++i)
#pragma unroll
    for (int j = 0; j < 4; ++j) acc[i][j] = z;

  stage(0, 0);
  wait_async0();
  __syncthreads();

  for (int kt = 0; kt < KT; ++kt) {
    const int cur = kt & 1;
    if (kt + 1 < KT) stage(kt + 1, cur ^ 1);

    v16h af[2], bf[4];
#pragma unroll
    for (int i = 0; i < 2; ++i)
      af[i] = load_frag16(&sA[cur][(wm * 32 + i * 16) * 32], 32, lane);
#pragma unroll
    for (int j = 0; j < 4; ++j)
      bf[j] = load_frag16(&sB[cur][(wn * 64 + j * 16) * 32], 32, lane);
#pragma unroll
    for (int i = 0; i < 2; ++i)
#pragma unroll
      for (int j = 0; j < 4; ++j) acc[i][j] = wmma16(af[i], bf[j], acc[i][j]);

    wait_async0();
    __syncthreads();
  }

  const int rl = (lane >> 4) << 3;
#pragma unroll
  for (int i = 0; i < 2; ++i) {
#pragma unroll
    for (int j = 0; j < 4; ++j) {
      const int gn = n0 + wn * 64 + j * 16 + (lane & 15);
      const float bn = bias[gn];
#pragma unroll
      for (int e = 0; e < 8; ++e) {
        const int gm = m0 + wm * 32 + i * 16 + rl + e;
        C[(size_t)gm * kDim + gn] = acc[i][j][e] + bn;
      }
    }
  }
}

// ---------------------------------------------------------------------------
// Host-side launcher.
// ---------------------------------------------------------------------------
extern "C" void kernel_launch(void* const* d_in, const int* in_sizes, int n_in,
                              void* d_out, int out_size, void* d_ws,
                              size_t ws_size, hipStream_t stream) {
  (void)in_sizes; (void)n_in; (void)out_size; (void)ws_size;
  const float* x    = (const float*)d_in[0];   // [2,2048,1024]
  const float* Wqkv = (const float*)d_in[1];   // [1024,3072]
  const float* bqkv = (const float*)d_in[2];   // [3072]
  const float* Wout = (const float*)d_in[3];   // [1024,1024]
  const float* bout = (const float*)d_in[4];   // [1024]
  float* out = (float*)d_out;                  // [2,2048,1024]

  char* ws = (char*)d_ws;
  size_t off = 0;
  auto alloc = [&](size_t bytes) -> void* {
    void* p = ws + off;
    off += (bytes + 255) & ~(size_t)255;
    return p;
  };
  const size_t nX = (size_t)kM * kDim;                 // 4.19M
  _Float16* xh    = (_Float16*)alloc(nX * 2);          // reused as Oh later
  _Float16* wqkvT = (_Float16*)alloc((size_t)3 * kDim * kDim * 2);
  _Float16* woutT = (_Float16*)alloc((size_t)kDim * kDim * 2);
  _Float16* qh    = (_Float16*)alloc(nX * 2);
  _Float16* kh    = (_Float16*)alloc(nX * 2);
  _Float16* vh    = (_Float16*)alloc(nX * 2);
  _Float16* oh    = xh;  // xh dead after QKV GEMM; alias for attention output

  // 1) precision/layout conversion (bandwidth-trivial: ~30 MB total)
  k_f32_to_f16<<<(int)((nX + 255) / 256), 256, 0, stream>>>(x, xh, (int)nX);
  k_transpose_f16<<<(kDim * 3 * kDim + 255) / 256, 256, 0, stream>>>(
      Wqkv, wqkvT, kDim, 3 * kDim);
  k_transpose_f16<<<(kDim * kDim + 255) / 256, 256, 0, stream>>>(
      Wout, woutT, kDim, kDim);

  // 2) QKV projection -> q/k/v in [B,H,N,Dh] f16
  k_qkv_gemm<<<dim3(kM / 128, (3 * kDim) / 128), 256, 0, stream>>>(
      xh, wqkvT, bqkv, qh, kh, vh);

  // 3) flash attention -> Oh in [B,N,C] f16
  k_attn<<<dim3(kSeq / 128, kB * kHeads), 256, 0, stream>>>(qh, kh, vh, oh);

  // 4) output projection -> f32 result
  k_out_gemm<<<dim3(kM / 128, kDim / 128), 256, 0, stream>>>(
      oh, woutT, bout, out);
}
